// AugementationAttention_63419487093457
// MI455X (gfx1250) — matrix operand order repaired and checked by reading, not compiled, not run on hardware
//
#include <hip/hip_runtime.h>

// ---------------------------------------------------------------------------
// AugmentationAttention for MI455X (gfx1250, wave32, WMMA + TDM).
// B=16, N=512, D=4096, NH=16, DK=256, 2D=8192.
// fp32 -> bf16 once, all GEMMs on v_wmma_f32_16x16x32_bf16 (f32 accum).
// Projection GEMM: 128x128 tile, K-step 64, double-buffered LDS filled by the
// Tensor Data Mover (tensor_load_to_lds + s_wait_tensorcnt) when available,
// manual b128 staging otherwise. Fused attention with LDS softmax.
// Workspace: ~608 MB (see carve() calls below).
// ---------------------------------------------------------------------------

#define B_    16
#define N_    512
#define D_    4096
#define NH_   16
#define DK_   256
#define TWO_D 8192
#define MROWS 8192   // B_*N_

typedef __attribute__((ext_vector_type(16))) __bf16         v16bf;
typedef __attribute__((ext_vector_type(8)))  float          v8f;
typedef __attribute__((ext_vector_type(4)))  unsigned int   u32x4;
typedef __attribute__((ext_vector_type(8)))  int            i32x8;
typedef __attribute__((ext_vector_type(4)))  int            i32x4;
typedef __attribute__((ext_vector_type(4)))  float          f32x4;
typedef __attribute__((ext_vector_type(4)))  unsigned short u16x4;

#if __has_builtin(__builtin_amdgcn_tensor_load_to_lds) && \
    __has_builtin(__builtin_amdgcn_s_wait_tensorcnt)
#define USE_TDM 1
#else
#define USE_TDM 0
#endif

union BF16Frag {
    v16bf          v;
    u32x4          q[2];
    unsigned short u[16];
};

__device__ __forceinline__ unsigned short f32_to_bf16_rne(float f) {
    unsigned int u = __float_as_uint(f);
    unsigned int r = u + 0x7FFFu + ((u >> 16) & 1u);
    return (unsigned short)(r >> 16);
}
__device__ __forceinline__ float bf16_to_f32(unsigned short h) {
    return __uint_as_float(((unsigned int)h) << 16);
}

// Generic (flat) LDS pointer -> workgroup-relative LDS byte offset.
// Per ISA 10.2 aperture mapping, flat-LDS addr[31:0] is the LDS offset.
__device__ __forceinline__ unsigned lds_byte_offset(const void* p) {
    return (unsigned)(unsigned long long)p;
}

#if USE_TDM
// ---------------------------------------------------------------------------
// Issue one TDM 2-D tile load (bf16 elements): tile_w x tile_h elements from a
// row-major tensor with row stride `row_stride` (elements) into contiguous LDS
// at byte offset ldsOff. D# packing per CDNA5 ISA 8.3/8.4.
// Must be executed by a single wave; tracked with TENSORcnt.
// ---------------------------------------------------------------------------
__device__ __forceinline__ void tdm_load_tile_2d(unsigned long long gaddr_bytes,
                                                 unsigned ldsOff,
                                                 unsigned tile_w, unsigned tile_h,
                                                 unsigned long long row_stride) {
    u32x4 g0;
    g0[0] = 1u;                                        // count=1, user descriptor
    g0[1] = ldsOff;                                    // lds_addr (bytes)
    g0[2] = (unsigned)(gaddr_bytes & 0xFFFFFFFFull);   // global_addr[31:0]
    g0[3] = (unsigned)((gaddr_bytes >> 32) & 0x01FFFFFFull) | (2u << 30);  // [56:32] | type=2

    const unsigned tw = (unsigned)row_stride;          // tensor_dim0: >= tile_w (no OOB)
    const unsigned th = 0x00100000u;                   // tensor_dim1: huge (no OOB)
    i32x8 g1;
    g1[0] = (int)(1u << 16);                                                // data_size=1 (2B), mask=0
    g1[1] = (int)((tw & 0xFFFFu) << 16);                                    // tensor_dim0[15:0]
    g1[2] = (int)(((tw >> 16) & 0xFFFFu) | ((th & 0xFFFFu) << 16));         // dim0[31:16] | dim1[15:0]
    g1[3] = (int)(((th >> 16) & 0xFFFFu) | ((tile_w & 0xFFFFu) << 16));     // dim1[31:16] | tile_dim0
    g1[4] = (int)(tile_h & 0xFFFFu);                                        // tile_dim1 (tile_dim2=0)
    g1[5] = (int)(row_stride & 0xFFFFFFFFull);                              // tensor_dim0_stride[31:0]
    g1[6] = (int)((row_stride >> 32) & 0xFFFFull);                          // stride[47:32], dim1_stride=0
    g1[7] = 0;

    i32x4 gz = {0, 0, 0, 0};
#if __clang_major__ >= 23
    i32x8 gz8 = {0, 0, 0, 0, 0, 0, 0, 0};
    __builtin_amdgcn_tensor_load_to_lds(g0, g1, gz, gz, gz8, 0);
#else
    __builtin_amdgcn_tensor_load_to_lds(g0, g1, gz, gz, 0);
#endif
}
#endif  // USE_TDM

// ---------------------------------------------------------------------------
// fp32 -> bf16 conversion (n always a multiple of 4 here)
// ---------------------------------------------------------------------------
__global__ __launch_bounds__(256) void cvt_f32_bf16(const float* __restrict__ src,
                                                    unsigned short* __restrict__ dst,
                                                    long long n) {
    long long i = ((long long)blockIdx.x * blockDim.x + threadIdx.x) * 4;
    if (i + 3 < n) {
        f32x4 f = *(const f32x4*)(src + i);
        u16x4 o;
        o.x = f32_to_bf16_rne(f.x);
        o.y = f32_to_bf16_rne(f.y);
        o.z = f32_to_bf16_rne(f.z);
        o.w = f32_to_bf16_rne(f.w);
        *(u16x4*)(dst + i) = o;
    }
}

// ---------------------------------------------------------------------------
// C[m][c] = sum_k A[m][k] * W[c][k]   (torch Linear: x @ W.T)
// Tile: 128 rows x 128 cols per 256-thread (8-wave) workgroup; K-step 64.
// Double-buffered LDS; TDM prefetch (wave 0) when available.
// MODE 0: EP   -> bf16 row-major [m][Nout], bias added
// MODE 1: Q/K  -> bf16 head-major [(b*NH+h)][seq][dk]
// MODE 2: V    -> bf16 transposed head-major [(b*NH+h)][dk][seq]
// ---------------------------------------------------------------------------
template <int MODE>
__global__ __launch_bounds__(256) void gemm_bt_wmma(const unsigned short* __restrict__ A,
                                                    const unsigned short* __restrict__ W,
                                                    const float* __restrict__ bias,
                                                    unsigned short* __restrict__ Out,
                                                    int K, int Nout) {
    __shared__ unsigned short Atile[2][128][64];  // 32 KB
    __shared__ unsigned short Wtile[2][128][64];  // 32 KB

    const int tid  = threadIdx.x;
    const int lane = tid & 31;
    const int wave = tid >> 5;
    const int rowbase = blockIdx.y * 128;
    const int colbase = blockIdx.x * 128;

    // fragment geometry (ISA 7.12.2)
    const int frow = lane & 15;
    const int ak0  = (lane & 16) ? 8 : 0;    // A frag: chunks at k0 and k0+16
    const int bk0  = (lane & 16) ? 8 : 0;    // B frag: chunks at k0 and k0+16
    const int mloc = (lane & 16) ? 8 : 0;

    v8f acc[8] = {};  // 8 n-tiles of 16 cols each
    const int NSTEP = K >> 6;  // K / 64

#if USE_TDM
    const unsigned long long aBase = (unsigned long long)(const void*)A +
                                     ((unsigned long long)rowbase * K) * 2ull;
    const unsigned long long wBase = (unsigned long long)(const void*)W +
                                     ((unsigned long long)colbase * K) * 2ull;
    const unsigned aLds[2] = {lds_byte_offset(&Atile[0][0][0]), lds_byte_offset(&Atile[1][0][0])};
    const unsigned wLds[2] = {lds_byte_offset(&Wtile[0][0][0]), lds_byte_offset(&Wtile[1][0][0])};
    if (wave == 0) {
        tdm_load_tile_2d(aBase, aLds[0], 64, 128, (unsigned long long)K);
        tdm_load_tile_2d(wBase, wLds[0], 64, 128, (unsigned long long)K);
    }
#else
    // manual staging assignments: 128x64 halves, 256 threads -> 32 halves each
    const int srow = tid >> 1;             // 0..127
    const int soff = (tid & 1) * 32;       // 0 or 32 halves
#endif

    for (int i = 0; i < NSTEP; ++i) {
        const int bufi = i & 1;
        const int kb = i << 6;

#if USE_TDM
        if (wave == 0) {
            if (i + 1 < NSTEP) {
                const unsigned long long koff = (unsigned long long)(kb + 64) * 2ull;
                tdm_load_tile_2d(aBase + koff, aLds[(i + 1) & 1], 64, 128, (unsigned long long)K);
                tdm_load_tile_2d(wBase + koff, wLds[(i + 1) & 1], 64, 128, (unsigned long long)K);
                __builtin_amdgcn_s_wait_tensorcnt(2);  // step i's two tiles done
            } else {
                __builtin_amdgcn_s_wait_tensorcnt(0);
            }
        }
        __syncthreads();
#else
        {
            const unsigned short* ga = &A[(size_t)(rowbase + srow) * K + kb + soff];
            const unsigned short* gw = &W[(size_t)(colbase + srow) * K + kb + soff];
#pragma unroll
            for (int c = 0; c < 4; ++c) {
                *(u32x4*)&Atile[bufi][srow][soff + c * 8] = *(const u32x4*)(ga + c * 8);
                *(u32x4*)&Wtile[bufi][srow][soff + c * 8] = *(const u32x4*)(gw + c * 8);
            }
        }
        __syncthreads();
#endif

#pragma unroll
        for (int kk = 0; kk < 64; kk += 32) {
            BF16Frag a;
            const unsigned short* ap = &Atile[bufi][wave * 16 + frow][kk + ak0];
            a.q[0] = *(const u32x4*)ap;
            a.q[1] = *(const u32x4*)(ap + 16);
#pragma unroll
            for (int t = 0; t < 8; ++t) {
                BF16Frag b;
                const unsigned short* wp = &Wtile[bufi][t * 16 + frow][kk + bk0];
                b.q[0] = *(const u32x4*)wp;
                b.q[1] = *(const u32x4*)(wp + 16);
                acc[t] = __builtin_amdgcn_wmma_f32_16x16x32_bf16(
                    false, a.v, false, b.v, (short)0, acc[t], false, false);
            }
        }
        __syncthreads();
    }

#pragma unroll
    for (int t = 0; t < 8; ++t) {
#pragma unroll
        for (int e = 0; e < 8; ++e) {
            const int gm = rowbase + wave * 16 + mloc + e;
            const int c  = colbase + t * 16 + frow;
            float vv = acc[t][e];
            if (MODE == 0) {
                vv += bias[c];
                Out[(size_t)gm * Nout + c] = f32_to_bf16_rne(vv);
            } else if (MODE == 1) {
                const int bidx = gm >> 9, seq = gm & 511;
                const int h = c >> 8, dk = c & 255;
                Out[(((size_t)bidx * NH_ + h) * N_ + seq) * DK_ + dk] = f32_to_bf16_rne(vv);
            } else {
                const int bidx = gm >> 9, seq = gm & 511;
                const int h = c >> 8, dk = c & 255;
                Out[(((size_t)bidx * NH_ + h) * DK_ + dk) * N_ + seq] = f32_to_bf16_rne(vv);
            }
        }
    }
}

// ---------------------------------------------------------------------------
// Fused attention: one 256-thread (8-wave) workgroup per (b, h, 16-row qtile).
//   S = (Q K^T) / 16 + EP     (each wave: 16x64 slab via 8x4 WMMAs)
//   softmax rows in LDS, P -> bf16
//   O = P V                   (each wave: 16x32 slab via 16x2 WMMAs)
// Q/K head-major [bh][seq][dk]; Vt transposed [bh][dk][seq]; EP bf16 [m][2D].
// Q/K/V fragments load straight from global (L2-resident working set).
// ---------------------------------------------------------------------------
__global__ __launch_bounds__(256) void attn_wmma(const unsigned short* __restrict__ Q,
                                                 const unsigned short* __restrict__ Km,
                                                 const unsigned short* __restrict__ Vt,
                                                 const unsigned short* __restrict__ EP,
                                                 float* __restrict__ Out) {
    __shared__ float          Sbuf[16][512];   // 32 KB
    __shared__ unsigned short Pbuf[16][512];   // 16 KB
    __shared__ float          red[16][16];     // 1 KB

    const int tid  = threadIdx.x;
    const int lane = tid & 31;
    const int wave = tid >> 5;
    const int qt = blockIdx.x;        // 0..31
    const int bh = blockIdx.y;        // 0..255
    const int b  = bh >> 4, h = bh & 15;
    const int qbase = qt * 16;

    const size_t headoff = (size_t)bh * N_ * DK_;
    const int frow = lane & 15;
    const int ak0  = (lane & 16) ? 8 : 0;
    const int bk0  = (lane & 16) ? 8 : 0;
    const int mloc = (lane & 16) ? 8 : 0;

    // cache all Q fragments (16 rows x 256 dk = 8 K-chunks) in registers
    BF16Frag qf[8];
#pragma unroll
    for (int kk = 0; kk < 8; ++kk) {
        const unsigned short* qp = Q + headoff + (size_t)(qbase + frow) * DK_ + kk * 32 + ak0;
        qf[kk].q[0] = *(const u32x4*)qp;
        qf[kk].q[1] = *(const u32x4*)(qp + 16);
    }

    const float scale = 0.0625f;  // 1/sqrt(256)

    // ---- S = Q K^T * scale + EP ; wave covers seq cols [wave*64, wave*64+64)
#pragma unroll
    for (int t = 0; t < 4; ++t) {
        const int colb = wave * 64 + t * 16;
        v8f acc = {};
#pragma unroll
        for (int kk = 0; kk < 8; ++kk) {
            BF16Frag kf;
            const unsigned short* kp =
                Km + headoff + (size_t)(colb + frow) * DK_ + kk * 32 + bk0;
            kf.q[0] = *(const u32x4*)kp;
            kf.q[1] = *(const u32x4*)(kp + 16);
            acc = __builtin_amdgcn_wmma_f32_16x16x32_bf16(
                false, qf[kk].v, false, kf.v, (short)0, acc, false, false);
        }
#pragma unroll
        for (int e = 0; e < 8; ++e) {
            const int m = mloc + e;
            const int n = colb + frow;
            const float ep =
                bf16_to_f32(EP[(size_t)(b * N_ + qbase + m) * TWO_D + h * N_ + n]);
            Sbuf[m][n] = acc[e] * scale + ep;
        }
    }
    __syncthreads();

    // ---- softmax over 512-wide rows: thread (r = tid/16, g = tid%16) owns 32 cols
    {
        const int r = tid >> 4, g = tid & 15;
        float mx = -1e30f;
        for (int j = 0; j < 32; ++j) mx = fmaxf(mx, Sbuf[r][g * 32 + j]);
        red[r][g] = mx;
        __syncthreads();
        float rowmax = red[r][0];
        for (int j = 1; j < 16; ++j) rowmax = fmaxf(rowmax, red[r][j]);
        __syncthreads();
        float s = 0.f;
        for (int j = 0; j < 32; ++j) {
            float e = __expf(Sbuf[r][g * 32 + j] - rowmax);
            Sbuf[r][g * 32 + j] = e;
            s += e;
        }
        red[r][g] = s;
        __syncthreads();
        float rowsum = 0.f;
        for (int j = 0; j < 16; ++j) rowsum += red[r][j];
        const float inv = 1.0f / rowsum;
        for (int j = 0; j < 32; ++j)
            Pbuf[r][g * 32 + j] = f32_to_bf16_rne(Sbuf[r][g * 32 + j] * inv);
    }
    __syncthreads();

    // ---- O = P V ; wave covers dk cols [wave*32, wave*32+32)
    const size_t vtoff = (size_t)bh * DK_ * N_;
    v8f oacc[2] = {};
    for (int kk = 0; kk < 16; ++kk) {  // seq chunks of 32
        BF16Frag pf;
        const unsigned short* pp = &Pbuf[frow][kk * 32 + ak0];
        pf.q[0] = *(const u32x4*)pp;
        pf.q[1] = *(const u32x4*)(pp + 16);
#pragma unroll
        for (int t = 0; t < 2; ++t) {
            BF16Frag vf;
            const unsigned short* vp =
                Vt + vtoff + (size_t)(wave * 32 + t * 16 + frow) * N_ + kk * 32 + bk0;
            vf.q[0] = *(const u32x4*)vp;
            vf.q[1] = *(const u32x4*)(vp + 16);
            oacc[t] = __builtin_amdgcn_wmma_f32_16x16x32_bf16(
                false, pf.v, false, vf.v, (short)0, oacc[t], false, false);
        }
    }
#pragma unroll
    for (int t = 0; t < 2; ++t) {
#pragma unroll
        for (int e = 0; e < 8; ++e) {
            const int m  = mloc + e;
            const int dk = wave * 32 + t * 16 + frow;
            Out[(size_t)(b * N_ + qbase + m) * D_ + h * DK_ + dk] = oacc[t][e];
        }
    }
}

// ---------------------------------------------------------------------------
extern "C" void kernel_launch(void* const* d_in, const int* in_sizes, int n_in,
                              void* d_out, int out_size, void* d_ws, size_t ws_size,
                              hipStream_t stream) {
    const float* x   = (const float*)d_in[0];
    const float* aug = (const float*)d_in[1];
    const float* Wq  = (const float*)d_in[2];
    const float* Wk  = (const float*)d_in[3];
    const float* Wv  = (const float*)d_in[4];
    const float* Wp  = (const float*)d_in[5];
    const float* bp  = (const float*)d_in[6];
    float* out = (float*)d_out;

    char* ws = (char*)d_ws;
    size_t off = 0;
    auto carve = [&](size_t bytes) {
        char* p = ws + off;
        off += bytes;
        return (unsigned short*)p;
    };
    unsigned short* xb   = carve((size_t)MROWS * D_ * 2);      // 64 MB
    unsigned short* augb = carve((size_t)MROWS * D_ * 2);      // 64 MB
    unsigned short* wqb  = carve((size_t)D_ * D_ * 2);         // 32 MB
    unsigned short* wkb  = carve((size_t)D_ * D_ * 2);         // 32 MB
    unsigned short* wvb  = carve((size_t)D_ * D_ * 2);         // 32 MB
    unsigned short* wpb  = carve((size_t)TWO_D * D_ * 2);      // 64 MB
    unsigned short* Qb   = carve((size_t)MROWS * D_ * 2);      // 64 MB
    unsigned short* Kb   = carve((size_t)MROWS * D_ * 2);      // 64 MB
    unsigned short* Vtb  = carve((size_t)MROWS * D_ * 2);      // 64 MB
    unsigned short* epb  = carve((size_t)MROWS * TWO_D * 2);   // 128 MB

    auto cvt = [&](const float* s, unsigned short* d, long long n) {
        unsigned blocks = (unsigned)((n / 4 + 255) / 256);
        cvt_f32_bf16<<<dim3(blocks), dim3(256), 0, stream>>>(s, d, n);
    };
    cvt(x,   xb,   (long long)MROWS * D_);
    cvt(aug, augb, (long long)MROWS * D_);
    cvt(Wq,  wqb,  (long long)D_ * D_);
    cvt(Wk,  wkb,  (long long)D_ * D_);
    cvt(Wv,  wvb,  (long long)D_ * D_);
    cvt(Wp,  wpb,  (long long)TWO_D * D_);

    // projections: grid = (Nout/128, M/128)
    gemm_bt_wmma<1><<<dim3(D_ / 128, MROWS / 128), dim3(256), 0, stream>>>(
        xb, wqb, nullptr, Qb, D_, D_);
    gemm_bt_wmma<1><<<dim3(D_ / 128, MROWS / 128), dim3(256), 0, stream>>>(
        xb, wkb, nullptr, Kb, D_, D_);
    gemm_bt_wmma<2><<<dim3(D_ / 128, MROWS / 128), dim3(256), 0, stream>>>(
        xb, wvb, nullptr, Vtb, D_, D_);
    gemm_bt_wmma<0><<<dim3(TWO_D / 128, MROWS / 128), dim3(256), 0, stream>>>(
        augb, wpb, bp, epb, D_, TWO_D);

    // fused attention: grid = (N/16 qtiles, B*NH heads)
    attn_wmma<<<dim3(N_ / 16, B_ * NH_), dim3(256), 0, stream>>>(Qb, Kb, Vtb, epb, out);
}
